// DPSOG_Water_47579647705261
// MI455X (gfx1250) — matrix operand order repaired and checked by reading, not compiled, not run on hardware
//
#include <hip/hip_runtime.h>
#include <hip/hip_bf16.h>
#include <math.h>

// ---------------------------------------------------------------------------
// Problem constants (match reference)
// ---------------------------------------------------------------------------
#define BB       4
#define NP       768
#define NO_      256
#define NH_      512
#define MA_TOT   42      // 14 + 28
#define MR_TOT   180     // 60 + 120
#define EDIM     32      // embedding output width
#define DDIM     160     // 4*32 + 32
#define NF       128     // fourier terms
#define NCH      4       // sog channels

typedef __attribute__((ext_vector_type(16))) _Float16     v16h;
typedef __attribute__((ext_vector_type(8)))  float        v8f;
typedef __attribute__((ext_vector_type(4)))  unsigned int v4u;
typedef __attribute__((ext_vector_type(8)))  int          v8i;
typedef __attribute__((ext_vector_type(4)))  int          v4i;

// ---------------------------------------------------------------------------
// Workspace layout (floats)
// ---------------------------------------------------------------------------
constexpr size_t SZ_GIA  = (size_t)BB * NP * MA_TOT * EDIM;
constexpr size_t SZ_RIA  = (size_t)BB * NP * MA_TOT * 4;
constexpr size_t SZ_DR   = (size_t)BB * NP * EDIM;
constexpr size_t SZ_D    = (size_t)BB * NP * DDIM;
constexpr size_t OFF_GIA = 0;
constexpr size_t OFF_RIA = OFF_GIA + SZ_GIA;
constexpr size_t OFF_DR  = OFF_RIA + SZ_RIA;
constexpr size_t OFF_D   = OFF_DR  + SZ_DR;
constexpr size_t OFF_Q   = OFF_D   + SZ_D;

// ---------------------------------------------------------------------------
// WMMA helper (wave32, gfx1250)
// ---------------------------------------------------------------------------
__device__ __forceinline__ v8f wmma_f16(v16h a, v16h b, v8f c) {
  // D(16x16,f32) = A(16x32,f16) * B(32x16,f16) + C
  return __builtin_amdgcn_wmma_f32_16x16x32_f16(
      /*neg_a=*/false, a, /*neg_b=*/false, b,
      /*c_mod=*/(short)0, c, /*reuse_a=*/false, /*reuse_b=*/false);
}

// C/D fragment store: lane L holds col N=L&15, rows M=(L>>4)*8 + r.
__device__ __forceinline__ void store_cd(float* C, int ldn, int nOff, v8f d) {
  int lane = threadIdx.x & 31;
  int n    = lane & 15;
  int mb   = (lane >> 4) * 8;
#pragma unroll
  for (int r = 0; r < 8; ++r) C[(mb + r) * ldn + nOff + n] = d[r];
}

// Fragment-major LDS addressing (per CDNA5 ISA 7.12.2 layouts).
// A fragment (16x32 f16): element (row, kk) -> lane l, half h:
//   l = row + ((kk>>3)&1)*16 ; v = ((kk>>4)&1)*4 + ((kk>>1)&3) ; h = 2v + (kk&1)
__device__ __forceinline__ void afrag_put(v16h* frag, int row, int kk,
                                          _Float16 val) {
  int l = row + (((kk >> 3) & 1) << 4);
  int v = (((kk >> 4) & 1) << 2) + ((kk >> 1) & 3);
  int h = 2 * v + (kk & 1);
  ((_Float16*)&frag[l])[h] = val;
}
// B fragment (32x16 f16): element (kk, n) -> lane l = ((kk>>3)&1)*16 + n, h = ((kk>>4)&1)*8 + (kk&7)
__device__ __forceinline__ void bfrag_put(v16h* frag, int kk, int n,
                                          _Float16 val) {
  int l = (((kk >> 3) & 1) << 4) + n;
  int h = (((kk >> 4) & 1) << 3) + (kk & 7);
  ((_Float16*)&frag[l])[h] = val;
}

// ---------------------------------------------------------------------------
// Tensor Data Mover: 1-D tile, 4-byte elements, global -> LDS (ISA ch.8).
// Group0: count=1 | lds_addr | global_addr(57b) | type=2.
// Group1: data_size=2(4B); tensor_dim0=n; tensor_dim1=1; tile_dim0=n;
//         tile_dim1=1; stride0=n.  Groups 2/3 zero (<=2D tensor).
// ---------------------------------------------------------------------------
__device__ __forceinline__ void tdm_load_linear(unsigned lds_off,
                                                const void* g,
                                                unsigned n_dwords) {
  unsigned long long ga = (unsigned long long)(size_t)g;
  v4u g0 = {1u, lds_off, (unsigned)ga,
            (unsigned)((ga >> 32) & 0x01ffffffu) | 0x80000000u};
  v8i g1 = {(int)(2u << 16),                                  // data_size=4B
            (int)((n_dwords & 0xffffu) << 16),                // dim0[15:0]
            (int)(((n_dwords >> 16) & 0xffffu) | (1u << 16)), // dim0[31:16]|dim1=1
            (int)((n_dwords & 0xffffu) << 16),                // tile_dim0=n
            1,                                                // tile_dim1=1
            (int)n_dwords,                                    // stride0
            0, 0};
  v4i z = {0, 0, 0, 0};
#if __has_include(<hip/amd_detail/amd_gfx1250_TDM.h>)
  v8i z8 = {0, 0, 0, 0, 0, 0, 0, 0};
  __builtin_amdgcn_tensor_load_to_lds(g0, g1, z, z, z8, 0);
#else
  __builtin_amdgcn_tensor_load_to_lds(g0, g1, z, z, 0);
#endif
}

__device__ __forceinline__ unsigned lds_offset_of(const void* p) {
  // generic->LDS: low 32 bits of a shared-aperture address are the LDS offset
  return (unsigned)(size_t)p;
}

// ---------------------------------------------------------------------------
// Embedding MLP: dims [1,2,4,8,16,32], tanh, residual (dout==2*din -> +[x,x])
// ---------------------------------------------------------------------------
struct MLP5 {
  const float* W[5];
  const float* b[5];
};

__device__ __forceinline__ void embed_mlp(float s, const MLP5 P, float* o) {
  float x1[2];
#pragma unroll
  for (int j = 0; j < 2; ++j)
    x1[j] = tanhf(fmaf(s, P.W[0][j], P.b[0][j])) + s;
  float x2[4];
#pragma unroll
  for (int j = 0; j < 4; ++j) {
    float t = P.b[1][j];
#pragma unroll
    for (int i = 0; i < 2; ++i) t = fmaf(x1[i], P.W[1][i * 4 + j], t);
    x2[j] = tanhf(t) + x1[j & 1];
  }
  float x3[8];
#pragma unroll
  for (int j = 0; j < 8; ++j) {
    float t = P.b[2][j];
#pragma unroll
    for (int i = 0; i < 4; ++i) t = fmaf(x2[i], P.W[2][i * 8 + j], t);
    x3[j] = tanhf(t) + x2[j & 3];
  }
  float x4[16];
#pragma unroll
  for (int j = 0; j < 16; ++j) {
    float t = P.b[3][j];
#pragma unroll
    for (int i = 0; i < 8; ++i) t = fmaf(x3[i], P.W[3][i * 16 + j], t);
    x4[j] = tanhf(t) + x3[j & 7];
  }
#pragma unroll
  for (int j = 0; j < 32; ++j) {
    float t = P.b[4][j];
#pragma unroll
    for (int i = 0; i < 16; ++i) t = fmaf(x4[i], P.W[4][i * 32 + j], t);
    o[j] = tanhf(t) + x4[j & 15];
  }
}

// ---------------------------------------------------------------------------
// K0: zero a buffer
// ---------------------------------------------------------------------------
__global__ void k_zero(float* p, int n) {
  for (int t = blockIdx.x * blockDim.x + threadIdx.x; t < n;
       t += gridDim.x * blockDim.x)
    p[t] = 0.f;
}

// ---------------------------------------------------------------------------
// K1: per-pair envelope + embedding.
// mode 0 (A): store GI (32f) and RI (4f); mode 1 (R): atomicAdd into Dr.
// ---------------------------------------------------------------------------
__global__ void k_env_embed(const float* __restrict__ pos,
                            const int* __restrict__ idx,
                            const float* __restrict__ Lp,
                            const float* __restrict__ innerP,
                            const float* __restrict__ rcutP,
                            int nCenters, int centerOff, int neighOff,
                            int M, int mOff, MLP5 P, int mode,
                            float* __restrict__ GI, float* __restrict__ RI,
                            float* __restrict__ Dr) {
  int t = blockIdx.x * blockDim.x + threadIdx.x;
  int total = BB * nCenters * M;
  if (t >= total) return;
  int m = t % M;
  int n = (t / M) % nCenters;
  int b = t / (M * nCenters);

  float L    = Lp[0];
  float rcut = rcutP[0];
  float rin  = innerP[0] * rcut;

  const float* pc = pos + ((size_t)b * NP + centerOff + n) * 3;
  int jn = idx[((size_t)b * nCenters + n) * M + m];
  const float* pn = pos + ((size_t)b * NP + neighOff + jn) * 3;

  float dx = pn[0] - pc[0];
  float dy = pn[1] - pc[1];
  float dz = pn[2] - pc[2];
  dx -= L * roundf(dx / L);
  dy -= L * roundf(dy / L);
  dz -= L * roundf(dz / L);
  float r   = sqrtf(dx * dx + dy * dy + dz * dz + 1e-12f);
  float inv = 1.f / r;
  float u   = (r - rin) / (rcut - rin);
  float poly = u * u * u * (-6.f * u * u + 15.f * u - 10.f) + 1.f;
  float s = (r < rin) ? inv : ((r < rcut) ? inv * poly : 0.f);
  if (r < 1e-4f) s = 0.f;

  float g[32];
  embed_mlp(s, P, g);

  int row = centerOff + n;
  if (mode == 0) {
    size_t base = (((size_t)b * NP + row) * MA_TOT + (mOff + m)) * EDIM;
#pragma unroll
    for (int c = 0; c < 32; ++c) GI[base + c] = g[c];
    size_t rb = (((size_t)b * NP + row) * MA_TOT + (mOff + m)) * 4;
    RI[rb + 0] = s;
    RI[rb + 1] = s * inv * dx;
    RI[rb + 2] = s * inv * dy;
    RI[rb + 3] = s * inv * dz;
  } else {
    size_t base = ((size_t)b * NP + row) * EDIM;
#pragma unroll
    for (int c = 0; c < 32; ++c) atomicAdd(&Dr[base + c], g[c]);
  }
}

// ---------------------------------------------------------------------------
// K2: per-atom descriptor. One wave per atom.
// TDM stages GI (42x32 f32) and RI (42x4 f32) into LDS asynchronously;
// lanes convert into fragment-major f16 buffers; RiaGia via 4x WMMA;
// Da Gram product VALU; D row assembled.
// ---------------------------------------------------------------------------
__global__ void k_descriptor(const float* __restrict__ GI_A,
                             const float* __restrict__ RI_A,
                             const float* __restrict__ Dr,
                             float* __restrict__ D) {
  __shared__ float stageGI[MA_TOT * EDIM];   // TDM landing (f32)
  __shared__ float stageRI[MA_TOT * 4];      // TDM landing (f32)
  __shared__ v16h  Afrag[2][32];             // A = RI^T, 2 K-chunks
  __shared__ v16h  Bfrag[2][2][32];          // [kchunk][ntile][lane]
  __shared__ float RG[16 * 32];              // RiaGia (rows f, cols d)

  int lane = threadIdx.x;
  size_t atom = blockIdx.x;                  // 0 .. B*NP-1

  // Kick off async DMA of both operand tiles (wave-level, uniform address).
  tdm_load_linear(lds_offset_of(stageGI), GI_A + atom * MA_TOT * EDIM,
                  MA_TOT * EDIM);
  tdm_load_linear(lds_offset_of(stageRI), RI_A + atom * MA_TOT * 4,
                  MA_TOT * 4);

  // Overlap: zero fragment buffers (covers K padding 42..63 and rows 4..15).
  v16h z = {};
#pragma unroll
  for (int c = 0; c < 2; ++c) {
    Afrag[c][lane] = z;
    Bfrag[c][0][lane] = z;
    Bfrag[c][1][lane] = z;
  }

  __builtin_amdgcn_s_wait_tensorcnt(0);
  __syncthreads();

  // RI^T -> A fragments (rows f=0..3, K=m)
  for (int e = lane; e < MA_TOT * 4; e += 32) {
    int m = e >> 2, f = e & 3;
    afrag_put(Afrag[m >> 5], f, m & 31, (_Float16)stageRI[e]);
  }
  // GI -> B fragments (K=m, N=d)
  for (int e = lane; e < MA_TOT * EDIM; e += 32) {
    int m = e >> 5, d = e & 31;
    bfrag_put(Bfrag[m >> 5][d >> 4], m & 31, d & 15, (_Float16)stageGI[e]);
  }
  __syncthreads();

  v8f acc0 = {}, acc1 = {};
#pragma unroll
  for (int c = 0; c < 2; ++c) {
    v16h a = Afrag[c][lane];
    acc0 = wmma_f16(a, Bfrag[c][0][lane], acc0);
    acc1 = wmma_f16(a, Bfrag[c][1][lane], acc1);
  }
  store_cd(RG, 32, 0, acc0);
  store_cd(RG, 32, 16, acc1);
  __syncthreads();

  const float sc = 1.f / (float)(MA_TOT * MA_TOT);
  float* Drow = D + atom * DDIM;
#pragma unroll
  for (int t = 0; t < 4; ++t) {
    int i = lane * 4 + t;                    // 128 Da entries, 4 per lane
    int s = i >> 5, d = i & 31;
    float v = 0.f;
#pragma unroll
    for (int f = 0; f < 4; ++f) v += RG[f * 32 + s] * RG[f * 32 + d];
    Drow[i] = v * sc;
  }
  Drow[128 + lane] = Dr[atom * EDIM + lane] * (1.f / (float)MR_TOT);
}

// ---------------------------------------------------------------------------
// K4: fit + charge nets, one wave per 16-atom tile.
// TDM stages the 16x160 D tile; 160->16 front layers of both nets via WMMA
// (K=160 in 5 chunks); tails per-lane VALU.
// ---------------------------------------------------------------------------
__global__ void k_fit(const float* __restrict__ D,
                      const float* Wf0, const float* bf0,
                      const float* Wf1, const float* bf1,
                      const float* Wf2, const float* bf2,
                      const float* Wf3, const float* bf3,
                      const float* Wf4, const float* bf4,
                      const float* Wl,  const float* bl,
                      const float* Wq0, const float* bq0,
                      const float* Wq1, const float* bq1,
                      const float* Wq2, const float* bq2,
                      float* __restrict__ qout, float* __restrict__ Eout) {
  __shared__ float stageD[16 * DDIM];        // TDM landing (f32)
  __shared__ v16h  AfragD[5][32];
  __shared__ v16h  BfragF[5][32];
  __shared__ v16h  BfragQ[5][32];
  __shared__ float X1[16 * 16];
  __shared__ float XQ[16 * 16];

  int lane = threadIdx.x;
  size_t a0 = (size_t)blockIdx.x * 16;

  // Async DMA of the activation tile (16 atoms x 160 feats, contiguous).
  tdm_load_linear(lds_offset_of(stageD), D + a0 * DDIM, 16 * DDIM);

  // Overlap DMA with weight conversion into fragment-major layout.
  for (int e = lane; e < DDIM * 16; e += 32) {
    int K = e >> 4, n = e & 15;
    bfrag_put(BfragF[K >> 5], K & 31, n, (_Float16)Wf0[e]);
    bfrag_put(BfragQ[K >> 5], K & 31, n, (_Float16)Wq0[e]);
  }

  __builtin_amdgcn_s_wait_tensorcnt(0);
  __syncthreads();

  for (int e = lane; e < 16 * DDIM; e += 32) {
    int a = e / DDIM, K = e % DDIM;
    afrag_put(AfragD[K >> 5], a, K & 31, (_Float16)stageD[e]);
  }
  __syncthreads();

  v8f af = {}, aq = {};
#pragma unroll
  for (int c = 0; c < 5; ++c) {
    v16h A = AfragD[c][lane];
    af = wmma_f16(A, BfragF[c][lane], af);
    aq = wmma_f16(A, BfragQ[c][lane], aq);
  }
  store_cd(X1, 16, 0, af);
  store_cd(XQ, 16, 0, aq);
  __syncthreads();

  if (lane < 16) {
    size_t arow = a0 + lane;
    int b = (int)(arow / NP);

    float x[16];
#pragma unroll
    for (int j = 0; j < 16; ++j) x[j] = tanhf(X1[lane * 16 + j] + bf0[j]);
    float y8[8];
#pragma unroll
    for (int j = 0; j < 8; ++j) {
      float t = bf1[j];
#pragma unroll
      for (int i = 0; i < 16; ++i) t = fmaf(x[i], Wf1[i * 8 + j], t);
      y8[j] = tanhf(t);
    }
    float y4[4];
#pragma unroll
    for (int j = 0; j < 4; ++j) {
      float t = bf2[j];
#pragma unroll
      for (int i = 0; i < 8; ++i) t = fmaf(y8[i], Wf2[i * 4 + j], t);
      y4[j] = tanhf(t);
    }
    float y2[2];
#pragma unroll
    for (int j = 0; j < 2; ++j) {
      float t = bf3[j];
#pragma unroll
      for (int i = 0; i < 4; ++i) t = fmaf(y4[i], Wf3[i * 2 + j], t);
      y2[j] = tanhf(t);
    }
    float t1 = bf4[0];
#pragma unroll
    for (int i = 0; i < 2; ++i) t1 = fmaf(y2[i], Wf4[i], t1);
    float fo = tanhf(t1);
    atomicAdd(&Eout[b], fo * Wl[0] + bl[0]);

    // Q net tail: 16->16 (+res) -> 16 (+res); q = first 4 channels
    float z1[16];
#pragma unroll
    for (int j = 0; j < 16; ++j) z1[j] = tanhf(XQ[lane * 16 + j] + bq0[j]);
    float z2[16];
#pragma unroll
    for (int j = 0; j < 16; ++j) {
      float t = bq1[j];
#pragma unroll
      for (int i = 0; i < 16; ++i) t = fmaf(z1[i], Wq1[i * 16 + j], t);
      z2[j] = tanhf(t) + z1[j];
    }
    float z3[16];
#pragma unroll
    for (int j = 0; j < 16; ++j) {
      float t = bq2[j];
#pragma unroll
      for (int i = 0; i < 16; ++i) t = fmaf(z2[i], Wq2[i * 16 + j], t);
      z3[j] = tanhf(t) + z2[j];
    }
#pragma unroll
    for (int c = 0; c < NCH; ++c) qout[arow * NCH + c] = z3[c];
  }
}

// ---------------------------------------------------------------------------
// K5: long-range sum-of-Gaussians, fused energy + direct force term.
// Per (b, i): E += 0.5 * sum_{j!=i} sum_k w_k.(q_i*q_j) exp(-a_k r2_ij)
// F_i       = -sum_{j!=i} 2 * [sum_k a_k w_k.(q_i*q_j) exp(-a_k r2)] * d_ij
// Transcendental-bound: r2 recomputed from LDS tiles, alpha-loop fused.
// ---------------------------------------------------------------------------
__global__ void k_sog(const float* __restrict__ pos,
                      const float* __restrict__ qv,
                      const float* __restrict__ alpha,
                      const float* __restrict__ w,      // [4][128]
                      const float* __restrict__ Lp,
                      float* __restrict__ Eout, float* __restrict__ F) {
  __shared__ float sA[NF];
  __shared__ float sW[NCH * NF];
  __shared__ float sP[64 * 3];
  __shared__ float sQ[64 * NCH];

  int tid = threadIdx.x;
  int b = blockIdx.x / 3;
  int i = (blockIdx.x % 3) * 256 + tid;
  float L = Lp[0];

  for (int e = tid; e < NF; e += 256) sA[e] = alpha[e];
  for (int e = tid; e < NCH * NF; e += 256) sW[e] = w[e];

  const float* pi = pos + ((size_t)b * NP + i) * 3;
  float px = pi[0], py = pi[1], pz = pi[2];
  const float* qi = qv + ((size_t)b * NP + i) * NCH;
  float q0 = qi[0], q1 = qi[1], q2 = qi[2], q3 = qi[3];

  float esum = 0.f, fx = 0.f, fy = 0.f, fz = 0.f;

  for (int jc = 0; jc < NP / 64; ++jc) {
    __syncthreads();
    if (tid < 64) {
      int j = jc * 64 + tid;
      const float* pj = pos + ((size_t)b * NP + j) * 3;
      sP[tid * 3 + 0] = pj[0];
      sP[tid * 3 + 1] = pj[1];
      sP[tid * 3 + 2] = pj[2];
      const float* qj = qv + ((size_t)b * NP + j) * NCH;
      sQ[tid * 4 + 0] = qj[0];
      sQ[tid * 4 + 1] = qj[1];
      sQ[tid * 4 + 2] = qj[2];
      sQ[tid * 4 + 3] = qj[3];
      if (jc + 1 < NP / 64) {   // prefetch next tile -> global_prefetch_b8
        __builtin_prefetch(pos + ((size_t)b * NP + (jc + 1) * 64 + tid) * 3, 0, 1);
        __builtin_prefetch(qv  + ((size_t)b * NP + (jc + 1) * 64 + tid) * NCH, 0, 1);
      }
    }
    __syncthreads();

    for (int jl = 0; jl < 64; ++jl) {
      int j = jc * 64 + jl;
      if (j == i) continue;
      float dx = sP[jl * 3 + 0] - px;
      float dy = sP[jl * 3 + 1] - py;
      float dz = sP[jl * 3 + 2] - pz;
      dx -= L * roundf(dx / L);
      dy -= L * roundf(dy / L);
      dz -= L * roundf(dz / L);
      float r2 = dx * dx + dy * dy + dz * dz;
      float P0 = q0 * sQ[jl * 4 + 0];
      float P1 = q1 * sQ[jl * 4 + 1];
      float P2 = q2 * sQ[jl * 4 + 2];
      float P3 = q3 * sQ[jl * 4 + 3];
      float eacc = 0.f, facc = 0.f;
#pragma unroll 4
      for (int k = 0; k < NF; ++k) {
        float al = sA[k];
        float e  = __expf(-al * r2);
        float wd = sW[k] * P0 + sW[NF + k] * P1 + sW[2 * NF + k] * P2 +
                   sW[3 * NF + k] * P3;
        float g = e * wd;
        eacc += g;
        facc = fmaf(al, g, facc);
      }
      esum += eacc;
      float fm = -2.f * facc;
      fx = fmaf(fm, dx, fx);
      fy = fmaf(fm, dy, fy);
      fz = fmaf(fm, dz, fz);
    }
  }

  atomicAdd(&Eout[b], 0.5f * esum);
  float* fo = F + ((size_t)b * NP + i) * 3;
  fo[0] = fx;
  fo[1] = fy;
  fo[2] = fz;
}

// ---------------------------------------------------------------------------
// Host-side launch. Input pointer indices follow setup_inputs() dict order
// with nested pytrees flattened in insertion order (see round-0 notes).
// ---------------------------------------------------------------------------
static MLP5 make_mlp5(void* const* d_in, int base) {
  MLP5 p;
  for (int l = 0; l < 5; ++l) {
    p.W[l] = (const float*)d_in[base + 2 * l];
    p.b[l] = (const float*)d_in[base + 2 * l + 1];
  }
  return p;
}

extern "C" void kernel_launch(void* const* d_in, const int* in_sizes, int n_in,
                              void* d_out, int out_size, void* d_ws,
                              size_t ws_size, hipStream_t stream) {
  (void)in_sizes; (void)n_in; (void)out_size; (void)ws_size;

  const float* pos   = (const float*)d_in[0];
  const float* Lp    = (const float*)d_in[2];
  const float* innA  = (const float*)d_in[4];
  const float* radA  = (const float*)d_in[5];
  const float* innR  = (const float*)d_in[10];
  const float* radR  = (const float*)d_in[11];
  const int* IdxA[4] = {(const int*)d_in[6], (const int*)d_in[8],
                        (const int*)d_in[7], (const int*)d_in[9]};
  const int* IdxR[4] = {(const int*)d_in[12], (const int*)d_in[14],
                        (const int*)d_in[13], (const int*)d_in[15]};

  MLP5 embA[4] = {make_mlp5(d_in, 16), make_mlp5(d_in, 26),
                  make_mlp5(d_in, 36), make_mlp5(d_in, 46)};
  MLP5 embR[4] = {make_mlp5(d_in, 56), make_mlp5(d_in, 66),
                  make_mlp5(d_in, 76), make_mlp5(d_in, 86)};

  float* ws   = (float*)d_ws;
  float* GIA  = ws + OFF_GIA;
  float* RIA  = ws + OFF_RIA;
  float* Dr   = ws + OFF_DR;
  float* Dd   = ws + OFF_D;
  float* qv   = ws + OFF_Q;
  float* Eout = (float*)d_out;          // [B]
  float* Fout = (float*)d_out + BB;     // [B][NP][3]

  // zero accumulators (energy, forces, Dr)
  k_zero<<<64, 256, 0, stream>>>(Eout, BB + BB * NP * 3);
  k_zero<<<(int)((SZ_DR + 255) / 256), 256, 0, stream>>>(Dr, (int)SZ_DR);

  // env + embedding combos: {nCenters, centerOff, neighOff, M, mOff}
  const int cA[4][5] = {{NO_, 0, 0, 14, 0},       {NO_, 0, NO_, 28, 14},
                        {NH_, NO_, 0, 14, 0},     {NH_, NO_, NO_, 28, 14}};
  const int cR[4][5] = {{NO_, 0, 0, 60, 0},       {NO_, 0, NO_, 120, 60},
                        {NH_, NO_, 0, 60, 0},     {NH_, NO_, NO_, 120, 60}};
  for (int c = 0; c < 4; ++c) {
    int total = BB * cA[c][0] * cA[c][3];
    k_env_embed<<<(total + 255) / 256, 256, 0, stream>>>(
        pos, IdxA[c], Lp, innA, radA, cA[c][0], cA[c][1], cA[c][2], cA[c][3],
        cA[c][4], embA[c], /*mode=*/0, GIA, RIA, Dr);
  }
  for (int c = 0; c < 4; ++c) {
    int total = BB * cR[c][0] * cR[c][3];
    k_env_embed<<<(total + 255) / 256, 256, 0, stream>>>(
        pos, IdxR[c], Lp, innR, radR, cR[c][0], cR[c][1], cR[c][2], cR[c][3],
        cR[c][4], embR[c], /*mode=*/1, GIA, RIA, Dr);
  }

  // descriptors: one wave per atom (TDM staging + WMMA for RI^T @ GI)
  k_descriptor<<<BB * NP, 32, 0, stream>>>(GIA, RIA, Dr, Dd);

  // fit + charge nets: one wave per 16-atom tile (TDM + WMMA 160->16 layers)
  k_fit<<<BB * NP / 16, 32, 0, stream>>>(
      Dd,
      (const float*)d_in[102], (const float*)d_in[103],
      (const float*)d_in[104], (const float*)d_in[105],
      (const float*)d_in[106], (const float*)d_in[107],
      (const float*)d_in[108], (const float*)d_in[109],
      (const float*)d_in[110], (const float*)d_in[111],
      (const float*)d_in[112], (const float*)d_in[113],
      (const float*)d_in[96],  (const float*)d_in[97],
      (const float*)d_in[98],  (const float*)d_in[99],
      (const float*)d_in[100], (const float*)d_in[101],
      qv, Eout);

  // long-range SOG energy + direct forces
  k_sog<<<BB * 3, 256, 0, stream>>>(pos, qv, (const float*)d_in[115],
                                    (const float*)d_in[114], Lp, Eout, Fout);
}
